// SwappingMechanism_24154896073077
// MI455X (gfx1250) — compile-verified, matrix operand order checked by reading, not run on hardware
//
#include <hip/hip_runtime.h>
#include <hip/hip_fp16.h>
#include <stdint.h>

#define G     64
#define GS    4096
#define FD    256
#define KSEL  2048
#define EPSF  1e-8f

typedef __attribute__((ext_vector_type(2)))  float    v2f;
typedef __attribute__((ext_vector_type(4)))  float    v4f;
typedef __attribute__((ext_vector_type(8)))  float    v8f;
typedef __attribute__((ext_vector_type(16))) _Float16 v16h;

#if defined(__has_builtin)
#if __has_builtin(__builtin_amdgcn_wmma_f32_16x16x4_f32)
#define HAVE_WMMA_F32X4 1
#endif
#endif

// ---------------- deterministic keyed permutation (Feistel on 12 bits) ----
static __device__ __forceinline__ uint32_t mix32(uint32_t x) {
    x ^= x >> 16; x *= 0x7feb352du;
    x ^= x >> 15; x *= 0x846ca68bu;
    x ^= x >> 16;
    return x;
}

static __device__ __forceinline__ int feistel12(int i, uint32_t key) {
    uint32_t L = (uint32_t)(i >> 6) & 63u;
    uint32_t R = (uint32_t)i & 63u;
#pragma unroll
    for (int r = 0; r < 4; ++r) {
        uint32_t F  = mix32(R + key + (uint32_t)r * 0x9E3779B9u) & 63u;
        uint32_t nL = R;
        R = L ^ F;
        L = nL;
    }
    return (int)((L << 6) | R);
}

__global__ void __launch_bounds__(256)
perm_kernel(int* __restrict__ p1, int* __restrict__ p2) {
    const int g = blockIdx.x;
    const uint32_t k1 = (uint32_t)(2 * g + 0) * 0x85EBCA6Bu + 0xC2B2AE35u;
    const uint32_t k2 = (uint32_t)(2 * g + 1) * 0x85EBCA6Bu + 0xC2B2AE35u;
    for (int i = threadIdx.x; i < GS; i += 256) {
        p1[g * GS + i] = feistel12(i, k1);
        p2[g * GS + i] = feistel12(i, k2);
    }
}

// ---------------- similarity via WMMA (diag of Z1*Z2^T, Z1*Z1^T, Z2*Z2^T) --
static __device__ __forceinline__ float pick8(v8f c, int j) {
    float r = c[0];
    r = (j == 1) ? c[1] : r;
    r = (j == 2) ? c[2] : r;
    r = (j == 3) ? c[3] : r;
    r = (j == 4) ? c[4] : r;
    r = (j == 5) ? c[5] : r;
    r = (j == 6) ? c[6] : r;
    r = (j == 7) ? c[7] : r;
    return r;
}

__global__ void __launch_bounds__(256)
sim_kernel(const float* __restrict__ z, const int* __restrict__ p1,
           const int* __restrict__ p2, float* __restrict__ sim) {
    // 32 blocks per group, 8 waves per block, 16 rows per wave.
    const int g    = blockIdx.x >> 5;
    const int blk  = blockIdx.x & 31;
    const int wave = threadIdx.x >> 5;
    const int lane = threadIdx.x & 31;
    const int r0   = blk * 128 + wave * 16;
    const int m    = lane & 15;   // row within 16-row tile this lane loads
    const int hi   = lane >> 4;   // lane half selects K sub-block

    const int row = r0 + m;
    const int i1  = p1[g * GS + row];
    const int i2  = p2[g * GS + row];
    const float* z1 = z + ((size_t)g * GS + (size_t)i1) * FD;
    const float* z2 = z + ((size_t)g * GS + (size_t)i2) * FD;

    v8f cdot = {}; v8f cn1 = {}; v8f cn2 = {};

#if defined(HAVE_WMMA_F32X4)
    // A 16x4 f32 layout: lane m holds K = 2*hi..2*hi+1 (2 VGPRs).
    // Same register image read as B (4x16) is the transpose -> wmma(a,a)=Z*Z^T.
#pragma unroll 4
    for (int kb = 0; kb < FD; kb += 4) {
        v2f a = *(const v2f*)(z1 + kb + 2 * hi);
        v2f b = *(const v2f*)(z2 + kb + 2 * hi);
        cdot = __builtin_amdgcn_wmma_f32_16x16x4_f32(false, a, false, b,
                                                     (short)0, cdot, false, false);
        cn1  = __builtin_amdgcn_wmma_f32_16x16x4_f32(false, a, false, a,
                                                     (short)0, cn1, false, false);
        cn2  = __builtin_amdgcn_wmma_f32_16x16x4_f32(false, b, false, b,
                                                     (short)0, cn2, false, false);
    }
#else
    // Fallback: f16 16x16x32 WMMA (codegen-confirmed builtin).
    for (int kb = 0; kb < FD; kb += 32) {
        v16h a, b;
#pragma unroll
        for (int v = 0; v < 8; ++v) {
            const int kk = (v < 4 ? 2 * v : 2 * v + 8) + (hi ? 8 : 0);
            float2 af = *(const float2*)(z1 + kb + kk);
            float2 bf = *(const float2*)(z2 + kb + kk);
            a[2 * v] = (_Float16)af.x; a[2 * v + 1] = (_Float16)af.y;
            b[2 * v] = (_Float16)bf.x; b[2 * v + 1] = (_Float16)bf.y;
        }
        cdot = __builtin_amdgcn_wmma_f32_16x16x32_f16(false, a, false, b,
                                                      (short)0, cdot, false, false);
        cn1  = __builtin_amdgcn_wmma_f32_16x16x32_f16(false, a, false, a,
                                                      (short)0, cn1, false, false);
        cn2  = __builtin_amdgcn_wmma_f32_16x16x32_f16(false, b, false, b,
                                                      (short)0, cn2, false, false);
    }
#endif

    // C/D 16x16 f32 layout: lanes 0-15 hold M=vgpr, N=lane;
    // lanes 16-31 hold M=vgpr+8, N=lane-16.  Diagonal (m,m) lives in:
    //   lanes 0..7  -> vgpr = lane,      m = lane
    //   lanes 24..31-> vgpr = lane - 24, m = lane - 16
    const int  j     = lane & 7;
    const bool valid = (lane < 8) || (lane >= 24);
    const int  drow  = hi ? (lane - 16) : lane;

    const float dot  = pick8(cdot, j);
    const float n1sq = pick8(cn1, j);
    const float n2sq = pick8(cn2, j);
    const float n1   = fmaxf(sqrtf(n1sq), EPSF);
    const float n2   = fmaxf(sqrtf(n2sq), EPSF);

    if (valid) {
        // Written once, read once by the sorter: stream past caches.
        __builtin_nontemporal_store(dot / (n1 * n2), &sim[g * GS + r0 + drow]);
    }
}

// ---------------- per-group bitonic argsort in LDS --------------------------
__global__ void __launch_bounds__(1024)
sort_kernel(const float* __restrict__ sim, int* __restrict__ sel) {
    __shared__ float key[GS];
    __shared__ int   idx[GS];
    const int g = blockIdx.x;

    for (int i = threadIdx.x; i < GS; i += 1024) {
        key[i] = -sim[g * GS + i];   // ascending sort of -sim == descending sim
        idx[i] = i;
    }
    __syncthreads();

    for (int k = 2; k <= GS; k <<= 1) {
        for (int j = k >> 1; j > 0; j >>= 1) {
            for (int i = threadIdx.x; i < GS; i += 1024) {
                const int ixj = i ^ j;
                if (ixj > i) {
                    const bool up = ((i & k) == 0);
                    const float ki = key[i], kx = key[ixj];
                    if (up ? (ki > kx) : (ki < kx)) {
                        key[i] = kx; key[ixj] = ki;
                        const int ti = idx[i]; idx[i] = idx[ixj]; idx[ixj] = ti;
                    }
                }
            }
            __syncthreads();
        }
    }

    for (int t = threadIdx.x; t < KSEL; t += 1024)
        sel[g * KSEL + t] = idx[t];
}

// ---------------- gather + scatter of selected rows (128-bit wide) ----------
__global__ void __launch_bounds__(256)
gather_kernel(const float* __restrict__ z, const int* __restrict__ p1,
              const int* __restrict__ p2, const int* __restrict__ sel,
              float* __restrict__ out) {
    // 2048 blocks per group; block handles one selected position j (4 rows).
    const int g = blockIdx.x >> 11;
    const int j = blockIdx.x & 2047;

    const int i  = sel[g * KSEL + j];
    const int r1 = p1[g * GS + i];
    const int r2 = p2[g * GS + i];

    const size_t gbase = (size_t)g * GS * FD;          // floats
    const v4f* z1 = (const v4f*)(z + gbase + (size_t)r1 * FD);
    const v4f* z2 = (const v4f*)(z + gbase + (size_t)r2 * FD);

    v4f* swp = (v4f*)out + (gbase >> 2);
    v4f* org = (v4f*)out + (((size_t)G * GS * FD + gbase) >> 2);

    const int which = threadIdx.x >> 6;   // 0..3 : which output row
    const int c     = threadIdx.x & 63;   // 128-bit chunk within row (FD/4 = 64)

    // 512 MB of write-once data: non-temporal stores keep it from rinsing
    // through the 192 MB L2 (which should retain the gathered z rows instead).
    if (which == 0) {
        __builtin_nontemporal_store(z2[c], &swp[(size_t)j * 64 + c]);          // swp[:K] = z2s
    } else if (which == 1) {
        __builtin_nontemporal_store(z1[c], &swp[(size_t)(j + KSEL) * 64 + c]); // swp[K:] = z1s
    } else if (which == 2) {
        __builtin_nontemporal_store(z1[c], &org[(size_t)j * 64 + c]);          // org[:K] = z1s
    } else {
        __builtin_nontemporal_store(z2[c], &org[(size_t)(j + KSEL) * 64 + c]); // org[K:] = z2s
    }
}

// ---------------- entry -----------------------------------------------------
extern "C" void kernel_launch(void* const* d_in, const int* in_sizes, int n_in,
                              void* d_out, int out_size, void* d_ws, size_t ws_size,
                              hipStream_t stream) {
    (void)in_sizes; (void)n_in; (void)out_size; (void)ws_size;
    const float* z = (const float*)d_in[0];
    // d_in[1] (labels y) is implicit in the layout; unused.

    int*   p1  = (int*)d_ws;                 // G*GS ints
    int*   p2  = p1 + (size_t)G * GS;        // G*GS ints
    float* sim = (float*)(p2 + (size_t)G * GS);  // G*GS floats
    int*   sel = (int*)(sim + (size_t)G * GS);   // G*KSEL ints
    float* out = (float*)d_out;

    perm_kernel  <<<G,        256,  0, stream>>>(p1, p2);
    sim_kernel   <<<G * 32,   256,  0, stream>>>(z, p1, p2, sim);
    sort_kernel  <<<G,        1024, 0, stream>>>(sim, sel);
    gather_kernel<<<G * 2048, 256,  0, stream>>>(z, p1, p2, sel, out);
}
